// TempMe_23235773071502
// MI455X (gfx1250) — compile-verified, matrix operand order checked by reading, not run on hardware
//
#include <hip/hip_runtime.h>
#include <math.h>

// ---------------------------------------------------------------------------
// CDNA5 (gfx1250) wave32 WMMA building blocks
// ---------------------------------------------------------------------------
typedef __attribute__((ext_vector_type(16))) __bf16 v16bf;
typedef __attribute__((ext_vector_type(8)))  float  v8f;

struct alignas(16) U128 { unsigned int x, y, z, w; };
union FragBF { v16bf v; U128 q[2]; };

__device__ __forceinline__ unsigned short f2bf(float f) {
  unsigned int u = __float_as_uint(f);
  u += 0x7FFFu + ((u >> 16) & 1u);   // round-to-nearest-even
  return (unsigned short)(u >> 16);
}

// Unguarded 16x32 bf16 operand fragment load (ISA 7.12.2 layout): the caller
// passes a per-lane pointer already positioned at
//   base + (tilerow + lane%16)*ld + (lane<16 ? 0 : 8)
// Lanes 0-15 pick up K={0..7,16..23}, lanes 16-31 K={8..15,24..31} as two
// 16-byte loads.  All row/col bounds are handled by zero-padding the buffers,
// so the hot loops are branch-free.
__device__ __forceinline__ v16bf ldfrag_ptr(const unsigned short* p) {
  FragBF f;
  f.q[0] = *reinterpret_cast<const U128*>(p);
  f.q[1] = *reinterpret_cast<const U128*>(p + 16);
  return f.v;
}

__device__ __forceinline__ v8f wmma_bf16(v16bf a, v16bf b, v8f c) {
  return __builtin_amdgcn_wmma_f32_16x16x32_bf16(false, a, false, b,
                                                 (short)0, c, false, false);
}

// ---------------------------------------------------------------------------
// Row norms + f32 -> bf16 token copy; rows >= nvalid are zero-filled so WMMA
// consumers never need bounds checks.  (block = one row of 768)
// ---------------------------------------------------------------------------
__global__ void k_rownorm_bf16(const float* __restrict__ X,
                               unsigned short* __restrict__ Xbf,
                               float* __restrict__ norms, int nvalid) {
  const int D = 768;
  int row = blockIdx.x, tid = threadIdx.x;
  unsigned short* br = Xbf + (size_t)row * D;
  if (row >= nvalid) {               // zero pad row (block-uniform branch)
    for (int e = tid; e < D; e += 256) br[e] = 0;
    if (tid == 0) norms[row] = 0.f;
    return;
  }
  const float* xr = X + (size_t)row * D;
  float ss = 0.f;
  for (int e = tid; e < D; e += 256) {
    float v = xr[e];
    ss += v * v;
    br[e] = f2bf(v);
  }
  for (int off = 16; off > 0; off >>= 1) ss += __shfl_down(ss, off, 32);
  __shared__ float part[8];
  if ((tid & 31) == 0) part[tid >> 5] = ss;
  __syncthreads();
  if (tid == 0) {
    float t = 0.f;
#pragma unroll
    for (int i = 0; i < 8; ++i) t += part[i];
    norms[row] = sqrtf(t);
  }
}

__global__ void k_f2bf(const float* __restrict__ X,
                       unsigned short* __restrict__ Y, int n) {
  int i = blockIdx.x * blockDim.x + threadIdx.x;
  int stride = gridDim.x * blockDim.x;
  for (; i < n; i += stride) Y[i] = f2bf(X[i]);
}

// ---------------------------------------------------------------------------
// Top-2 of cosine similarity per row.  n is always a multiple of 16 here, so
// all loads are unguarded.  One wave = one 16-row tile; two column tiles per
// A fragment; butterfly top-2 within each 16-lane half; diagonal included.
// ---------------------------------------------------------------------------
__global__ void k_top2(const unsigned short* __restrict__ tbf,
                       const float* __restrict__ norms, int n,
                       int* __restrict__ idx) {
  const int D = 768;
  int lane = threadIdx.x & 31;
  int wave = threadIdx.x >> 5;
  int ntiles = n >> 4;
  int tile = blockIdx.x * 8 + wave;
  if (tile >= ntiles) return;
  int rowbase = tile << 4;
  int lane_r = lane & 15;
  int lane_k = (lane & 16) ? 8 : 0;
  int hi = (lane & 16) ? 8 : 0;

  const unsigned short* arow = tbf + (size_t)(rowbase + lane_r) * D + lane_k;

  float nr[8];
#pragma unroll
  for (int v = 0; v < 8; ++v) nr[v] = norms[rowbase + v + hi];

  float rv1[8], rv2[8];
  int   ri1[8], ri2[8];
#pragma unroll
  for (int v = 0; v < 8; ++v) {
    rv1[v] = -__builtin_huge_valf(); rv2[v] = -__builtin_huge_valf();
    ri1[v] = 0; ri2[v] = 0;
  }

  auto process = [&](v8f acc, int colbase) {
    float nc = norms[colbase + lane_r];
#pragma unroll
    for (int v = 0; v < 8; ++v) {
      // fast reciprocal: cosine value only feeds an argmax
      float s = acc[v] * __builtin_amdgcn_rcpf(fmaxf(nr[v] * nc, 1e-8f));
      float w1 = s, w2 = -__builtin_huge_valf();
      int   j1 = colbase + lane_r, j2 = 0;
#pragma unroll
      for (int off = 1; off < 16; off <<= 1) {
        float p1 = __shfl_xor(w1, off, 32);
        int  pj1 = __shfl_xor(j1, off, 32);
        float p2 = __shfl_xor(w2, off, 32);
        int  pj2 = __shfl_xor(j2, off, 32);
        if (p1 > w1) {
          float nv2 = (w1 > p2) ? w1 : p2;
          int   nj2 = (w1 > p2) ? j1 : pj2;
          w1 = p1; j1 = pj1; w2 = nv2; j2 = nj2;
        } else if (p1 > w2) {
          w2 = p1; j2 = pj1;
        }
      }
      if (w1 > rv1[v]) {
        float nv2 = (rv1[v] > w2) ? rv1[v] : w2;
        int   nj2 = (rv1[v] > w2) ? ri1[v] : j2;
        rv1[v] = w1; ri1[v] = j1; rv2[v] = nv2; ri2[v] = nj2;
      } else if (w1 > rv2[v]) {
        rv2[v] = w1; ri2[v] = j1;
      }
    }
  };

  int jt = 0;
  for (; jt + 2 <= ntiles; jt += 2) {
    const unsigned short* b0 =
        tbf + (size_t)((jt << 4) + lane_r) * D + lane_k;
    const unsigned short* b1 = b0 + (size_t)16 * D;
    v8f acc0 = {}, acc1 = {};
#pragma unroll 4
    for (int kt = 0; kt < 24; ++kt) {
      v16bf a = ldfrag_ptr(arow + kt * 32);
      acc0 = wmma_bf16(a, ldfrag_ptr(b0 + kt * 32), acc0);
      acc1 = wmma_bf16(a, ldfrag_ptr(b1 + kt * 32), acc1);
    }
    process(acc0, jt << 4);
    process(acc1, (jt << 4) + 16);
  }
  if (jt < ntiles) {   // odd tail (e.g. n=2000 -> 125 tiles)
    const unsigned short* b0 =
        tbf + (size_t)((jt << 4) + lane_r) * D + lane_k;
    v8f acc0 = {};
#pragma unroll 4
    for (int kt = 0; kt < 24; ++kt) {
      v16bf a = ldfrag_ptr(arow + kt * 32);
      acc0 = wmma_bf16(a, ldfrag_ptr(b0 + kt * 32), acc0);
    }
    process(acc0, jt << 4);
  }

  if (lane_r == 0) {
#pragma unroll
    for (int v = 0; v < 8; ++v) {
      int row = rowbase + v + hi;
      idx[2 * row + 0] = ri1[v];
      idx[2 * row + 1] = ri2[v];
    }
  }
}

// ---------------------------------------------------------------------------
// Sequential merge scan (faithful to the lax.scan: later steps see earlier
// writes).  Single block; one barrier per step.
// ---------------------------------------------------------------------------
__global__ void k_merge_scan(float* __restrict__ tok,
                             const int* __restrict__ idx, int num_merge,
                             float wa, float wb) {
  const int D = 768;
  int tid = threadIdx.x;
  for (int i = 0; i < num_merge; ++i) {
    int a = idx[2 * i], b = idx[2 * i + 1];
    float* ra = tok + (size_t)a * D;
    float* rb = tok + (size_t)b * D;
    for (int e = tid; e < D; e += 256) {
      float nv = wa * ra[e] + wb * rb[e];
      ra[e] = nv;
      rb[e] = nv;
    }
    __syncthreads();
  }
}

// ---------------------------------------------------------------------------
// Y = X * W^T + bias (optional exact GELU).  X: [>=Lp x K] bf16 (rows >= L are
// zero), W: [N x K] bf16.  One 16x32 output tile per wave; branch-free K loop;
// epilogue zero-fills rows [L, Lp) so downstream WMMA consumers stay
// unguarded.
// ---------------------------------------------------------------------------
__global__ void k_gemm_bt(const unsigned short* __restrict__ Xbf,
                          const unsigned short* __restrict__ Wbf,
                          const float* __restrict__ bias,
                          float* __restrict__ Yf,
                          unsigned short* __restrict__ Ybf,
                          int L, int Lp, int N, int K, int do_gelu) {
  int lane = threadIdx.x & 31;
  int wave = threadIdx.x >> 5;
  int ntl = Lp >> 4;
  int ntn2 = N >> 5;
  int tile = blockIdx.x * 8 + wave;
  if (tile >= ntl * ntn2) return;
  int tl = tile / ntn2;
  int tn = tile - tl * ntn2;
  int rowbase = tl << 4, colbase = tn << 5;
  int lane_r = lane & 15;
  int lane_k = (lane & 16) ? 8 : 0;

  const unsigned short* pa = Xbf + (size_t)(rowbase + lane_r) * K + lane_k;
  const unsigned short* pb0 = Wbf + (size_t)(colbase + lane_r) * K + lane_k;
  const unsigned short* pb1 = pb0 + (size_t)16 * K;

  v8f acc0 = {}, acc1 = {};
  int nk = K >> 5;
#pragma unroll 4
  for (int kt = 0; kt < nk; ++kt) {
    v16bf a = ldfrag_ptr(pa);
    __builtin_prefetch(pa + 32, 0, 1);           // global_prefetch_b8
    acc0 = wmma_bf16(a, ldfrag_ptr(pb0), acc0);
    acc1 = wmma_bf16(a, ldfrag_ptr(pb1), acc1);
    pa += 32; pb0 += 32; pb1 += 32;
  }

  int col0 = colbase + lane_r;
  int hi = (lane & 16) ? 8 : 0;
  float bi0 = bias ? bias[col0] : 0.f;
  float bi1 = bias ? bias[col0 + 16] : 0.f;
#pragma unroll
  for (int v = 0; v < 8; ++v) {
    int row = rowbase + v + hi;
    size_t o = (size_t)row * N + col0;
    if (row < L) {
      float v0 = acc0[v] + bi0;
      float v1 = acc1[v] + bi1;
      if (do_gelu) {
        v0 = 0.5f * v0 * (1.f + erff(v0 * 0.70710678118f));
        v1 = 0.5f * v1 * (1.f + erff(v1 * 0.70710678118f));
      }
      if (Yf)  { Yf[o] = v0;       Yf[o + 16] = v1; }
      if (Ybf) { Ybf[o] = f2bf(v0); Ybf[o + 16] = f2bf(v1); }
    } else {                       // zero-pad rows [L, Lp)
      if (Yf)  { Yf[o] = 0.f; Yf[o + 16] = 0.f; }
      if (Ybf) { Ybf[o] = 0;  Ybf[o + 16] = 0; }
    }
  }
}

// ---------------------------------------------------------------------------
// Build V^T (per-head column-major V) in bf16: vt[h*96+c][key] = V[key][h*96+c]
// Keys >= L are zero.  2-D grid avoids per-element integer division.
// ---------------------------------------------------------------------------
__global__ void k_build_vt(const float* __restrict__ qkv,
                           unsigned short* __restrict__ vt, int L, int Lpad) {
  int c = blockIdx.y;                          // 0..767
  int key = blockIdx.x * 256 + threadIdx.x;    // 0..Lpad-1
  if (key >= Lpad) return;
  float v = (key < L) ? qkv[(size_t)key * 2304 + 1536 + c] : 0.f;
  vt[(size_t)c * Lpad + key] = f2bf(v);
}

// ---------------------------------------------------------------------------
// Flash-style attention: one wave per (head, 16-query tile).  qkvbf rows are
// zero-padded to Lpad (32-aligned), vt keys zero-padded, so all fragment
// loads are branch-free.  S = Q*K^T via WMMA, online softmax (masked cols get
// -inf), P re-fragmented through 1KB LDS, O += P*V via WMMA.
// ---------------------------------------------------------------------------
__global__ void k_attn(const unsigned short* __restrict__ qkvbf,
                       const unsigned short* __restrict__ vt,
                       unsigned short* __restrict__ out, int L, int Lpad) {
  const int DH = 96, QKVLD = 2304;
  int lane = threadIdx.x & 31;
  int qt = blockIdx.x;
  int h = blockIdx.y;
  int qbase = qt << 4;
  int lane_r = lane & 15;
  int lane_k = (lane & 16) ? 8 : 0;
  int hi = (lane & 16) ? 8 : 0;
  const float scale = 0.1020620726f;  // 1/sqrt(96)

  v16bf qf[3];
#pragma unroll
  for (int kt = 0; kt < 3; ++kt)
    qf[kt] = ldfrag_ptr(qkvbf + (size_t)(qbase + lane_r) * QKVLD +
                        h * DH + kt * 32 + lane_k);

  v8f vzero = {};
  v8f o[6];
#pragma unroll
  for (int t = 0; t < 6; ++t) o[t] = vzero;
  float m[8], l[8];
#pragma unroll
  for (int v = 0; v < 8; ++v) { m[v] = -__builtin_huge_valf(); l[v] = 0.f; }

  __shared__ unsigned short sP[16 * 32];

  int nkb = (L + 31) >> 5;
  for (int jb = 0; jb < nkb; ++jb) {
    int kb = jb << 5;
    const unsigned short* pk0 =
        qkvbf + (size_t)(kb + lane_r) * QKVLD + 768 + h * DH + lane_k;
    const unsigned short* pk1 = pk0 + (size_t)16 * QKVLD;
    v8f s0 = {}, s1 = {};
#pragma unroll
    for (int kt = 0; kt < 3; ++kt) {
      s0 = wmma_bf16(qf[kt], ldfrag_ptr(pk0 + kt * 32), s0);
      s1 = wmma_bf16(qf[kt], ldfrag_ptr(pk1 + kt * 32), s1);
    }
    int c0 = kb + lane_r;
    int c1 = c0 + 16;
#pragma unroll
    for (int v = 0; v < 8; ++v) {
      float sv0 = (c0 < L) ? s0[v] * scale : -__builtin_huge_valf();
      float sv1 = (c1 < L) ? s1[v] * scale : -__builtin_huge_valf();
      float tmax = fmaxf(sv0, sv1);
#pragma unroll
      for (int off = 1; off < 16; off <<= 1)
        tmax = fmaxf(tmax, __shfl_xor(tmax, off, 32));
      float mnew = fmaxf(m[v], tmax);
      float alpha = __expf(m[v] - mnew);
      float p0 = __expf(sv0 - mnew);
      float p1 = __expf(sv1 - mnew);
      float ps = p0 + p1;
#pragma unroll
      for (int off = 1; off < 16; off <<= 1)
        ps += __shfl_xor(ps, off, 32);
      l[v] = l[v] * alpha + ps;
      m[v] = mnew;
#pragma unroll
      for (int t = 0; t < 6; ++t) o[t][v] *= alpha;
      int r = v + hi;
      sP[r * 32 + lane_r]      = f2bf(p0);
      sP[r * 32 + 16 + lane_r] = f2bf(p1);
    }
    __syncthreads();
    v16bf pa = ldfrag_ptr(sP + lane_r * 32 + lane_k);
#pragma unroll
    for (int t = 0; t < 6; ++t) {
      const unsigned short* pv =
          vt + (size_t)(h * DH + t * 16 + lane_r) * Lpad + kb + lane_k;
      o[t] = wmma_bf16(pa, ldfrag_ptr(pv), o[t]);
    }
    __syncthreads();
  }

  int Lp16 = (L + 15) & ~15;
#pragma unroll
  for (int v = 0; v < 8; ++v) {
    int row = qbase + v + hi;
    size_t ob = (size_t)row * 768 + h * DH + lane_r;
    if (row < L) {
      float inv = __builtin_amdgcn_rcpf(l[v]);   // softmax normalization
#pragma unroll
      for (int t = 0; t < 6; ++t) out[ob + t * 16] = f2bf(o[t][v] * inv);
    } else if (row < Lp16) {       // keep pad rows zero for downstream GEMM
#pragma unroll
      for (int t = 0; t < 6; ++t) out[ob + t * 16] = 0;
    }
  }
}

// ---------------------------------------------------------------------------
// Y = LayerNorm(X + R) * g + b (block = one row)
// ---------------------------------------------------------------------------
__global__ void k_add_ln(const float* __restrict__ X, const float* __restrict__ R,
                         const float* __restrict__ g, const float* __restrict__ b,
                         float* __restrict__ Y) {
  const int D = 768;
  const float invD = 1.f / 768.f;
  int row = blockIdx.x, tid = threadIdx.x;
  __shared__ float z[768];
  __shared__ float red[16];
  __shared__ float mv[2];
  float s1 = 0.f, s2 = 0.f;
  for (int e = tid; e < D; e += 256) {
    float v = X[(size_t)row * D + e] + R[(size_t)row * D + e];
    z[e] = v;
    s1 += v;
    s2 += v * v;
  }
  for (int off = 16; off > 0; off >>= 1) {
    s1 += __shfl_down(s1, off, 32);
    s2 += __shfl_down(s2, off, 32);
  }
  if ((tid & 31) == 0) { red[tid >> 5] = s1; red[8 + (tid >> 5)] = s2; }
  __syncthreads();
  if (tid == 0) {
    float a = 0.f, c = 0.f;
#pragma unroll
    for (int i = 0; i < 8; ++i) { a += red[i]; c += red[8 + i]; }
    float mean = a * invD;
    float var = c * invD - mean * mean;
    mv[0] = mean;
    mv[1] = rsqrtf(var + 1e-5f);
  }
  __syncthreads();
  float mean = mv[0], inv = mv[1];
  for (int e = tid; e < D; e += 256)
    Y[(size_t)row * D + e] = (z[e] - mean) * inv * g[e] + b[e];
}

// ---------------------------------------------------------------------------
// Host orchestration
// ---------------------------------------------------------------------------
extern "C" void kernel_launch(void* const* d_in, const int* in_sizes, int n_in,
                              void* d_out, int out_size, void* d_ws,
                              size_t ws_size, hipStream_t stream) {
  (void)n_in; (void)out_size; (void)ws_size;
  const int D = 768;
  const float* vf   = (const float*)d_in[0];
  const float* Wqkv = (const float*)d_in[1];
  const float* bqkv = (const float*)d_in[2];
  const float* Wo   = (const float*)d_in[3];
  const float* bo   = (const float*)d_in[4];
  const float* g1   = (const float*)d_in[5];
  const float* b1g  = (const float*)d_in[6];
  const float* W1   = (const float*)d_in[7];
  const float* b1   = (const float*)d_in[8];
  const float* W2   = (const float*)d_in[9];
  const float* b2   = (const float*)d_in[10];
  const float* g2   = (const float*)d_in[11];
  const float* b2g  = (const float*)d_in[12];

  int n0 = in_sizes[0] / D;       // 8000
  int n1 = n0 - n0 / 2;           // 4000
  int n2 = n1 - n1 / 2;           // 2000
  int n3 = n2 - n2 / 4;           // 1500
  int Lpad = ((n2 + 31) / 32) * 32;          // 2016: key-padded QKV rows
  int n2p = (n2 + 15) & ~15;                 // 2000
  int n3p = (n3 + 15) & ~15;                 // 1504

  char* base = (char*)d_ws;
  size_t off = 0;
  auto carve = [&](size_t bytes) -> char* {
    char* p = base + off;
    off = (off + bytes + 255) & ~(size_t)255;
    return p;
  };
  float*          tokf  = (float*)carve((size_t)n0 * D * 4);
  unsigned short* tokbf = (unsigned short*)carve((size_t)n0 * D * 2);
  float*          norms = (float*)carve((size_t)n0 * 4);
  int*            idx   = (int*)carve((size_t)n0 * 2 * 4);
  unsigned short* wqkvb = (unsigned short*)carve((size_t)3 * D * D * 2);
  unsigned short* wob   = (unsigned short*)carve((size_t)D * D * 2);
  unsigned short* w1b   = (unsigned short*)carve((size_t)4 * D * D * 2);
  unsigned short* w2b   = (unsigned short*)carve((size_t)4 * D * D * 2);
  size_t big = (size_t)Lpad * 3 * D;                 // qkv (Lpad x 2304)
  if ((size_t)n3p * 4 * D > big) big = (size_t)n3p * 4 * D;  // H (n3p x 3072)
  float*          bigf  = (float*)carve(big * 4);
  unsigned short* bigbf = (unsigned short*)carve(big * 2);
  unsigned short* vtb   = (unsigned short*)carve((size_t)D * Lpad * 2);
  unsigned short* attnb = (unsigned short*)carve((size_t)n2p * D * 2);
  float*          projf = (float*)carve((size_t)n2p * D * 4);

  hipMemcpyAsync(tokf, vf, (size_t)n0 * D * 4, hipMemcpyDeviceToDevice, stream);

  k_f2bf<<<(3 * D * D + 255) / 256, 256, 0, stream>>>(Wqkv, wqkvb, 3 * D * D);
  k_f2bf<<<(D * D + 255) / 256, 256, 0, stream>>>(Wo, wob, D * D);
  k_f2bf<<<(4 * D * D + 255) / 256, 256, 0, stream>>>(W1, w1b, 4 * D * D);
  k_f2bf<<<(4 * D * D + 255) / 256, 256, 0, stream>>>(W2, w2b, 4 * D * D);

  int n = n0;
  for (int s = 0; s < 2; ++s) {   // two r=0.5 merges: 8000->4000->2000
    k_rownorm_bf16<<<n, 256, 0, stream>>>(tokf, tokbf, norms, n);
    int ntiles = n / 16;
    k_top2<<<(ntiles + 7) / 8, 256, 0, stream>>>(tokbf, norms, n, idx);
    int mg = n / 2;
    k_merge_scan<<<1, 256, 0, stream>>>(tokf, idx, mg, 0.5f, 0.5f);
    n -= mg;
  }

  // MHA on n2 tokens
  k_rownorm_bf16<<<n2p, 256, 0, stream>>>(tokf, tokbf, norms, n);
  {
    int nt = (Lpad / 16) * ((3 * D) / 32);
    k_gemm_bt<<<(nt + 7) / 8, 256, 0, stream>>>(tokbf, wqkvb, bqkv, bigf,
                                                bigbf, n, Lpad, 3 * D, D, 0);
  }
  k_build_vt<<<dim3((Lpad + 255) / 256, D), 256, 0, stream>>>(bigf, vtb, n,
                                                              Lpad);
  k_attn<<<dim3(n2p / 16, 8), 32, 0, stream>>>(bigbf, vtb, attnb, n, Lpad);
  {
    int nt = (n2p / 16) * (D / 32);
    k_gemm_bt<<<(nt + 7) / 8, 256, 0, stream>>>(attnb, wob, bo, projf, nullptr,
                                                n, n2p, D, D, 0);
  }
  k_add_ln<<<n, 256, 0, stream>>>(tokf, projf, g1, b1g, tokf);

  // third merge: r/2 = 0.25, weights 0.6 / 0.4
  k_rownorm_bf16<<<n, 256, 0, stream>>>(tokf, tokbf, norms, n);
  {
    int ntiles = n / 16;
    k_top2<<<(ntiles + 7) / 8, 256, 0, stream>>>(tokbf, norms, n, idx);
  }
  {
    int mg = n / 4;
    k_merge_scan<<<1, 256, 0, stream>>>(tokf, idx, mg, 0.6f, 0.4f);
    n -= mg;
  }

  // FFN on n3 tokens (1500 -> padded to 1504 rows)
  k_rownorm_bf16<<<n3p, 256, 0, stream>>>(tokf, tokbf, norms, n);
  {
    int nt = (n3p / 16) * ((4 * D) / 32);
    k_gemm_bt<<<(nt + 7) / 8, 256, 0, stream>>>(tokbf, w1b, b1, bigf, bigbf,
                                                n, n3p, 4 * D, D, 1);
  }
  {
    int nt = (n3p / 16) * (D / 32);
    k_gemm_bt<<<(nt + 7) / 8, 256, 0, stream>>>(bigbf, w2b, b2, projf, nullptr,
                                                n, n3p, D, 4 * D, 0);
  }
  k_add_ln<<<n, 256, 0, stream>>>(tokf, projf, g2, b2g, (float*)d_out);
}
